// SelfAttention_59193239274178
// MI455X (gfx1250) — matrix operand order repaired and checked by reading, not compiled, not run on hardware
//
#include <hip/hip_runtime.h>

typedef __attribute__((ext_vector_type(16))) _Float16 v16h;
typedef __attribute__((ext_vector_type(8)))  float    v8f;

#define CCH  256
#define NPOS 4096
#define BATCH 4

union Frag {
  v16h v;
  uint4 u[2];
};

__device__ __forceinline__ v8f wmma_f16(v16h a, v16h b, v8f c) {
  // D = A(16x32 f16) * B(32x16 f16) + C(16x16 f32)
  return __builtin_amdgcn_wmma_f32_16x16x32_f16(false, a, false, b, (short)0, c,
                                                false, false);
}

// ---------------------------------------------------------------------------
// GroupNorm: x[B,C,N] fp32 -> hT[B,N,C] f16  (normalize per (b, group of 32 ch))
// ---------------------------------------------------------------------------
__global__ __launch_bounds__(256) void gn_kernel(const float* __restrict__ x,
                                                 const float* __restrict__ gw,
                                                 const float* __restrict__ gb,
                                                 _Float16* __restrict__ hT) {
  const int b = blockIdx.x >> 3;
  const int g = blockIdx.x & 7;
  const float* xg = x + ((size_t)b * CCH + g * 32) * NPOS;
  float s = 0.f, ss = 0.f;
  for (int i = threadIdx.x; i < 32 * NPOS; i += 256) {
    float v = xg[i];
    s += v;
    ss += v * v;
  }
  __shared__ float rs[256], rq[256];
  rs[threadIdx.x] = s;
  rq[threadIdx.x] = ss;
  __syncthreads();
  for (int st = 128; st > 0; st >>= 1) {
    if (threadIdx.x < st) {
      rs[threadIdx.x] += rs[threadIdx.x + st];
      rq[threadIdx.x] += rq[threadIdx.x + st];
    }
    __syncthreads();
  }
  const float inv = 1.f / (32.f * NPOS);
  const float mean = rs[0] * inv;
  const float var = rq[0] * inv - mean * mean;
  const float rstd = rsqrtf(var + 1e-5f);
  for (int i = threadIdx.x; i < 32 * NPOS; i += 256) {
    const int cl = i >> 12;           // local channel (row of 4096)
    const int n = i & (NPOS - 1);
    const int c = g * 32 + cl;
    const float v = (xg[i] - mean) * rstd * gw[c] + gb[c];
    hT[((size_t)b * NPOS + n) * CCH + c] = (_Float16)v;
  }
}

// ---------------------------------------------------------------------------
// Weight fp32 -> f16 convert (256x256)
// ---------------------------------------------------------------------------
__global__ void wconv_kernel(const float* __restrict__ w, _Float16* __restrict__ wh) {
  const int i = blockIdx.x * 256 + threadIdx.x;
  wh[i] = (_Float16)w[i];
}

// ---------------------------------------------------------------------------
// q/k projection: outT[b][n][o] = (sum_c W[o][c] * hT[b][n][c] + bias[o]) * scale
// One wave -> one 16n x 16o tile.  A = hT rows (M=n), B = W rows (NN=o).
// scale = 1/sqrt(C) for Q (folds the attention scale), 1.0 for K.
// ---------------------------------------------------------------------------
__global__ __launch_bounds__(128) void gemm_qk_kernel(const _Float16* __restrict__ hT,
                                                      const _Float16* __restrict__ wh,
                                                      const float* __restrict__ bias,
                                                      _Float16* __restrict__ outT,
                                                      float scale) {
  const int lane = threadIdx.x & 31;
  const int wid  = threadIdx.x >> 5;
  const int lm   = lane & 15;
  const int hi   = lane >> 4;
  const int b    = blockIdx.z;
  const int n0   = blockIdx.x * 16;
  const int o0   = (blockIdx.y * 4 + wid) * 16;

  const _Float16* arow = hT + ((size_t)b * NPOS + n0 + lm) * CCH;
  const _Float16* brow = wh + (size_t)(o0 + lm) * CCH;
  v8f d = {0, 0, 0, 0, 0, 0, 0, 0};
#pragma unroll
  for (int kc = 0; kc < 8; ++kc) {
    Frag a, bb;
    const int ca = kc * 32 + (hi ? 16 : 0);
    a.u[0] = *(const uint4*)(arow + ca);
    a.u[1] = *(const uint4*)(arow + ca + 8);
    const int cb = kc * 32 + (hi ? 16 : 0);
    bb.u[0] = *(const uint4*)(brow + cb);
    bb.u[1] = *(const uint4*)(brow + cb + 8);
    d = wmma_f16(a.v, bb.v, d);
  }
  const float bo = bias[o0 + lm];
  _Float16* op = outT + ((size_t)b * NPOS + n0 + hi * 8) * CCH + o0 + lm;
#pragma unroll
  for (int j = 0; j < 8; ++j) op[(size_t)j * CCH] = (_Float16)((d[j] + bo) * scale);
}

// ---------------------------------------------------------------------------
// v projection: outN[b][o][n] = sum_c W[o][c] * hT[b][n][c] + bias[o]
// A = W rows (M=o), B = hT rows (NN=n).
// ---------------------------------------------------------------------------
__global__ __launch_bounds__(128) void gemm_v_kernel(const _Float16* __restrict__ hT,
                                                     const _Float16* __restrict__ wh,
                                                     const float* __restrict__ bias,
                                                     _Float16* __restrict__ outN) {
  const int lane = threadIdx.x & 31;
  const int wid  = threadIdx.x >> 5;
  const int lm   = lane & 15;
  const int hi   = lane >> 4;
  const int b    = blockIdx.z;
  const int n0   = blockIdx.x * 16;
  const int o0   = (blockIdx.y * 4 + wid) * 16;

  const _Float16* arow = wh + (size_t)(o0 + lm) * CCH;
  const _Float16* brow = hT + ((size_t)b * NPOS + n0 + lm) * CCH;
  v8f d = {0, 0, 0, 0, 0, 0, 0, 0};
#pragma unroll
  for (int kc = 0; kc < 8; ++kc) {
    Frag a, bb;
    const int ca = kc * 32 + (hi ? 16 : 0);
    a.u[0] = *(const uint4*)(arow + ca);
    a.u[1] = *(const uint4*)(arow + ca + 8);
    const int cb = kc * 32 + (hi ? 16 : 0);
    bb.u[0] = *(const uint4*)(brow + cb);
    bb.u[1] = *(const uint4*)(brow + cb + 8);
    d = wmma_f16(a.v, bb.v, d);
  }
  _Float16* op = outN + ((size_t)b * CCH + o0 + hi * 8) * NPOS + n0 + lm;
#pragma unroll
  for (int j = 0; j < 8; ++j)
    op[(size_t)j * NPOS] = (_Float16)(d[j] + bias[o0 + hi * 8 + j]);
}

// ---------------------------------------------------------------------------
// Flash attention.  Per wave: one 16-row query tile, full 256-ch O accumulator
// in registers (16 x v8f).  All 4 waves in the block share the 32-key K chunk,
// which is staged into LDS with CDNA5 async global->LDS DMA, double-buffered
// so the copy of chunk i+1 overlaps compute on chunk i.
//   S[m][n] = sum_c qT[m][c] * kT[n][c]      (scale pre-folded into qT)
//   O[c][m] = sum_n v[c][n] * P[m][n]  ->  oT[b][m][c] f16 (normalized)
// ---------------------------------------------------------------------------
__global__ __launch_bounds__(128) void attn_kernel(const _Float16* __restrict__ qT,
                                                   const _Float16* __restrict__ kT,
                                                   const _Float16* __restrict__ vN,
                                                   _Float16* __restrict__ oT) {
  __shared__ __align__(16) _Float16 kbuf[2][32][CCH];   // 2 x 16 KB
  __shared__ __align__(16) _Float16 pbuf[4][16][32];
  __shared__ float stats[4][16];
  const int tid  = threadIdx.x;
  const int lane = tid & 31;
  const int wid  = tid >> 5;
  const int lm   = lane & 15;
  const int hi   = lane >> 4;
  const int b    = blockIdx.y;
  const int m0   = (blockIdx.x * 4 + wid) * 16;

  const _Float16* qrow  = qT + ((size_t)b * NPOS + m0 + lm) * CCH;
  const _Float16* kbase = kT + (size_t)b * NPOS * CCH;
  const _Float16* vbase = vN + (size_t)b * CCH * NPOS;

  const v8f vzero = {0, 0, 0, 0, 0, 0, 0, 0};
  v8f acc[16];
#pragma unroll
  for (int i = 0; i < 16; ++i) acc[i] = vzero;
  float mrun[8], lrun[8];
#pragma unroll
  for (int r = 0; r < 8; ++r) { mrun[r] = -3.0e38f; lrun[r] = 0.f; }

  // --- async stage of one 32x256 f16 K chunk: 1024 x 16B, 8 per thread -----
  auto stage_k = [&](int nsrc, int buf) {
#pragma unroll
    for (int j = 0; j < 8; ++j) {
      const int e = tid + 128 * j;            // 0..1023 (16B units)
      const int row = e >> 5;                 // 32 x 16B per 512B row
      const int col = (e & 31) * 8;           // halfs
      const _Float16* g = kbase + (size_t)(nsrc + row) * CCH + col;
      const unsigned lds_off = (unsigned)(size_t)(&kbuf[buf][row][col]);
      asm volatile("global_load_async_to_lds_b128 %0, %1, off"
                   :: "v"(lds_off), "v"(g) : "memory");
    }
  };

  int cur = 0;
  stage_k(0, 0);
  asm volatile("s_wait_asynccnt 0x0" ::: "memory");
  __syncthreads();

  for (int n0 = 0; n0 < NPOS; n0 += 32) {
    if (n0 + 32 < NPOS) {
      stage_k(n0 + 32, cur ^ 1);              // overlap copy with compute
      // prefetch next V chunk (256 rows x 64B) into the cache hierarchy
#pragma unroll
      for (int r = 0; r < 8; ++r)
        __builtin_prefetch(vbase + (size_t)(r * 32 + lane) * NPOS + n0 + 32, 0, 3);
    }

    // ---- S tiles from LDS: s0 covers keys n0..n0+15, s1 covers n0+16..n0+31
    v8f s0 = vzero, s1 = vzero;
    const _Float16* krow0 = &kbuf[cur][lm][0];
    const _Float16* krow1 = &kbuf[cur][16 + lm][0];
#pragma unroll
    for (int kc = 0; kc < 8; ++kc) {
      Frag qa;
      const int ca = kc * 32 + (hi ? 16 : 0);
      qa.u[0] = *(const uint4*)(qrow + ca);
      qa.u[1] = *(const uint4*)(qrow + ca + 8);
      Frag b0, b1;
      const int cb = kc * 32 + (hi ? 16 : 0);
      b0.u[0] = *(const uint4*)(krow0 + cb);
      b0.u[1] = *(const uint4*)(krow0 + cb + 8);
      b1.u[0] = *(const uint4*)(krow1 + cb);
      b1.u[1] = *(const uint4*)(krow1 + cb + 8);
      s0 = wmma_f16(qa.v, b0.v, s0);
      s1 = wmma_f16(qa.v, b1.v, s1);
    }

    // ---- online softmax across this 32-key chunk.
    // D layout: VGPR r -> row m = r + 8*hi ; lane%16 -> key offset.
#pragma unroll
    for (int r = 0; r < 8; ++r) {
      const float a0 = s0[r];
      const float a1 = s1[r];
      float t = fmaxf(a0, a1);
      t = fmaxf(t, __shfl_xor(t, 1));
      t = fmaxf(t, __shfl_xor(t, 2));
      t = fmaxf(t, __shfl_xor(t, 4));
      t = fmaxf(t, __shfl_xor(t, 8));
      const float mnew = fmaxf(mrun[r], t);
      const float p0 = __expf(a0 - mnew);
      const float p1 = __expf(a1 - mnew);
      float rsum = p0 + p1;
      rsum += __shfl_xor(rsum, 1);
      rsum += __shfl_xor(rsum, 2);
      rsum += __shfl_xor(rsum, 4);
      rsum += __shfl_xor(rsum, 8);
      const float alpha = __expf(mrun[r] - mnew);
      lrun[r] = lrun[r] * alpha + rsum;
      mrun[r] = mnew;
      const int row = r + hi * 8;
      pbuf[wid][row][lm]      = (_Float16)p0;
      pbuf[wid][row][lm + 16] = (_Float16)p1;
      if (lm == r) stats[wid][row] = alpha;   // M-axis -> lane-axis transpose
    }
    asm volatile("s_wait_dscnt 0" ::: "memory");  // LDS in-order per wave; fence reads
    const float af = stats[wid][lm];

    // P as B fragment: lane -> m=lm, elements -> key offsets (hi selects 0..15/16..31)
    Frag pf;
    pf.u[0] = *(const uint4*)&pbuf[wid][lm][hi ? 16 : 0];
    pf.u[1] = *(const uint4*)&pbuf[wid][lm][(hi ? 16 : 0) + 8];

    // ---- O accumulate over all 16 channel tiles.  O-fragment lane -> m=lm.
#pragma unroll
    for (int ct = 0; ct < 16; ++ct) {
#pragma unroll
      for (int j = 0; j < 8; ++j) acc[ct][j] *= af;
      Frag vf;
      const _Float16* vrow = vbase + (size_t)(ct * 16 + lm) * NPOS + n0 + (hi ? 16 : 0);
      vf.u[0] = *(const uint4*)(vrow);
      vf.u[1] = *(const uint4*)(vrow + 8);
      acc[ct] = wmma_f16(vf.v, pf.v, acc[ct]);
    }

    // next-chunk K staging must be complete (own wave) and block-wide visible
    asm volatile("s_wait_asynccnt 0x0" ::: "memory");
    __syncthreads();
    cur ^= 1;
  }

  // ---- finalize: divide by softmax denominator, write oT[b][m][c] f16.
#pragma unroll
  for (int r = 0; r < 8; ++r)
    if (lm == r) stats[wid][r + hi * 8] = lrun[r];
  asm volatile("s_wait_dscnt 0" ::: "memory");
  const float linv = 1.f / stats[wid][lm];
  _Float16* orow = oT + ((size_t)b * NPOS + m0 + lm) * CCH;
#pragma unroll
  for (int ct = 0; ct < 16; ++ct) {
    union { _Float16 h[8]; uint4 u; } pk;
#pragma unroll
    for (int j = 0; j < 8; ++j) pk.h[j] = (_Float16)(acc[ct][j] * linv);
    *(uint4*)(orow + ct * 16 + hi * 8) = pk.u;
  }
}

// ---------------------------------------------------------------------------
// Projection + bias + residual: y[b][o][n] = x[b][o][n] + sum_c wp[o][c]*oT[b][n][c] + bp[o]
// ---------------------------------------------------------------------------
__global__ __launch_bounds__(128) void proj_kernel(const _Float16* __restrict__ oT,
                                                   const _Float16* __restrict__ wh,
                                                   const float* __restrict__ bias,
                                                   const float* __restrict__ x,
                                                   float* __restrict__ y) {
  const int lane = threadIdx.x & 31;
  const int wid  = threadIdx.x >> 5;
  const int lm   = lane & 15;
  const int hi   = lane >> 4;
  const int b    = blockIdx.z;
  const int n0   = blockIdx.x * 16;
  const int o0   = (blockIdx.y * 4 + wid) * 16;

  const _Float16* arow = wh + (size_t)(o0 + lm) * CCH;
  const _Float16* brow = oT + ((size_t)b * NPOS + n0 + lm) * CCH;
  v8f d = {0, 0, 0, 0, 0, 0, 0, 0};
#pragma unroll
  for (int kc = 0; kc < 8; ++kc) {
    Frag a, bb;
    const int ca = kc * 32 + (hi ? 16 : 0);
    a.u[0] = *(const uint4*)(arow + ca);
    a.u[1] = *(const uint4*)(arow + ca + 8);
    const int cb = kc * 32 + (hi ? 16 : 0);
    bb.u[0] = *(const uint4*)(brow + cb);
    bb.u[1] = *(const uint4*)(brow + cb + 8);
    d = wmma_f16(a.v, bb.v, d);
  }
  const size_t base = ((size_t)b * CCH + o0 + hi * 8) * NPOS + n0 + lm;
#pragma unroll
  for (int j = 0; j < 8; ++j)
    y[base + (size_t)j * NPOS] = x[base + (size_t)j * NPOS] + d[j] + bias[o0 + hi * 8 + j];
}

// ---------------------------------------------------------------------------
extern "C" void kernel_launch(void* const* d_in, const int* in_sizes, int n_in,
                              void* d_out, int out_size, void* d_ws, size_t ws_size,
                              hipStream_t stream) {
  (void)in_sizes; (void)n_in; (void)out_size; (void)ws_size;
  const float* x    = (const float*)d_in[0];
  const float* gn_w = (const float*)d_in[1];
  const float* gn_b = (const float*)d_in[2];
  const float* wq   = (const float*)d_in[3];
  const float* bq   = (const float*)d_in[4];
  const float* wk   = (const float*)d_in[5];
  const float* bk   = (const float*)d_in[6];
  const float* wv   = (const float*)d_in[7];
  const float* bv   = (const float*)d_in[8];
  const float* wp   = (const float*)d_in[9];
  const float* bp   = (const float*)d_in[10];
  float* y = (float*)d_out;

  char* p = (char*)d_ws;
  const size_t matBytes = (size_t)BATCH * NPOS * CCH * sizeof(_Float16);  // 8 MB
  _Float16* hT = (_Float16*)p; p += matBytes;
  _Float16* qT = (_Float16*)p; p += matBytes;
  _Float16* kT = (_Float16*)p; p += matBytes;
  _Float16* vN = (_Float16*)p; p += matBytes;
  _Float16* wqh = (_Float16*)p; p += (size_t)CCH * CCH * sizeof(_Float16);
  _Float16* wkh = (_Float16*)p; p += (size_t)CCH * CCH * sizeof(_Float16);
  _Float16* wvh = (_Float16*)p; p += (size_t)CCH * CCH * sizeof(_Float16);
  _Float16* wph = (_Float16*)p; p += (size_t)CCH * CCH * sizeof(_Float16);
  _Float16* oT = hT;  // hT is dead after the QKV GEMMs; reuse for attention output

  gn_kernel<<<32, 256, 0, stream>>>(x, gn_w, gn_b, hT);
  wconv_kernel<<<256, 256, 0, stream>>>(wq, wqh);
  wconv_kernel<<<256, 256, 0, stream>>>(wk, wkh);
  wconv_kernel<<<256, 256, 0, stream>>>(wv, wvh);
  wconv_kernel<<<256, 256, 0, stream>>>(wp, wph);

  gemm_qk_kernel<<<dim3(256, 4, BATCH), 128, 0, stream>>>(hT, wqh, bq, qT, 0.0625f);
  gemm_qk_kernel<<<dim3(256, 4, BATCH), 128, 0, stream>>>(hT, wkh, bk, kT, 1.0f);
  gemm_v_kernel <<<dim3(256, 4, BATCH), 128, 0, stream>>>(hT, wvh, bv, vN);

  attn_kernel<<<dim3(64, BATCH), 128, 0, stream>>>(qT, kT, vN, oT);

  proj_kernel<<<dim3(256, 4, BATCH), 128, 0, stream>>>(oT, wph, bp, x, y);
}
